// Model_45681272160983
// MI455X (gfx1250) — matrix operand order repaired and checked by reading, not compiled, not run on hardware
//
#include <hip/hip_runtime.h>
#include <hip/hip_bf16.h>
#include <stdint.h>

// Problem constants (from reference)
#define NN     100000
#define EE     400000
#define NDANG  4096
#define DD     512
#define FF     8000
#define EMB    300
#define HID    600      // 2*EMB
#define EMBP   320      // EMB padded to multiple of 32 (bf16 WMMA K)
#define HIDP   640      // HID padded to multiple of 32
#define NLAYERS 5

typedef __attribute__((ext_vector_type(16))) __bf16 v16bf;
typedef __attribute__((ext_vector_type(8)))  float  v8f;
typedef __attribute__((ext_vector_type(4)))  unsigned int v4u;

union Frag {
    v16bf          bf;
    unsigned short u[16];
    unsigned int   d[8];
    v4u            q[2];
};

__device__ __forceinline__ unsigned short f2bf(float f) {
    unsigned int u = __builtin_bit_cast(unsigned int, f);
    u += 0x7FFFu + ((u >> 16) & 1u);   // round-to-nearest-even
    return (unsigned short)(u >> 16);
}

// ---------------------------------------------------------------- embeddings
__global__ void embed_kernel(const int* __restrict__ x,
                             const float* __restrict__ ae1,
                             const float* __restrict__ ae2,
                             float* __restrict__ h) {
    int idx = blockIdx.x * blockDim.x + threadIdx.x;
    if (idx >= NN * EMB) return;
    int i = idx / EMB, c = idx - i * EMB;
    h[idx] = ae1[(size_t)x[2 * i] * EMB + c] + ae2[(size_t)x[2 * i + 1] * EMB + c];
}

// ------------------------------------------------- weight conversion (bf16, padded)
__global__ void conv_w_kernel(const float* __restrict__ W1,
                              const float* __restrict__ W2,
                              unsigned short* __restrict__ w1b,
                              unsigned short* __restrict__ w2b, int l) {
    const int S = EMBP * HIDP; // 204800
    int idx = blockIdx.x * blockDim.x + threadIdx.x;
    if (idx < S) {
        int k = idx / HIDP, n = idx - k * HIDP;
        float v = (k < EMB && n < HID)
                      ? W1[(size_t)l * EMB * HID + (size_t)k * HID + n] : 0.f;
        w1b[idx] = f2bf(v);
    } else if (idx < 2 * S) {
        int j = idx - S;
        int k = j / EMBP, n = j - k * EMBP;
        float v = (k < HID && n < EMB)
                      ? W2[(size_t)l * HID * EMB + (size_t)k * EMB + n] : 0.f;
        w2b[j] = f2bf(v);
    }
}

// ---------------------------------------------------------------- GIN aggregation
__global__ void aggr_init_kernel(const float* __restrict__ h,
                                 const float* __restrict__ ge1,
                                 const float* __restrict__ ge2,
                                 float* __restrict__ aggr, int l) {
    int idx = blockIdx.x * blockDim.x + threadIdx.x;
    if (idx >= NN * EMB) return;
    int c = idx % EMB;
    // self-loop edge_attr = (4, 0)
    aggr[idx] = h[idx] + ge1[((size_t)l * 6 + 4) * EMB + c]
                       + ge2[((size_t)l * 3 + 0) * EMB + c];
}

__global__ void scatter_kernel(const float* __restrict__ h,
                               const int* __restrict__ ei,
                               const int* __restrict__ ea,
                               const float* __restrict__ ge1,
                               const float* __restrict__ ge2,
                               float* aggr, int l) {
    long long idx = (long long)blockIdx.x * blockDim.x + threadIdx.x;
    if (idx >= (long long)EE * EMB) return;
    int e = (int)(idx / EMB), c = (int)(idx - (long long)e * EMB);
    int s  = ei[e], t = ei[EE + e];
    int a0 = ea[2 * e], a1 = ea[2 * e + 1];
    float v = h[(size_t)s * EMB + c]
            + ge1[((size_t)l * 6 + a0) * EMB + c]
            + ge2[((size_t)l * 3 + a1) * EMB + c];
    atomicAdd(&aggr[(size_t)t * EMB + c], v);
}

// ---------------------------------------------------------- fused GIN MLP on WMMA
// hpre = (relu(aggr @ W1 + b1)) @ W2 + b2, z kept in LDS as bf16.
// Block: 256 threads = 8 waves, tile 32 rows. Wave grid 2(M) x 4(N) of 16x16.
// B staging tile is stored COLUMN-major (bsh[n*40 + k]) so each lane's 16
// K values (two runs of 8 contiguous K) load as two 16B ds_load_b128 ops.
__global__ __launch_bounds__(256)
void mlp_wmma_kernel(const float* aggr,
                     const unsigned short* __restrict__ w1b,   // [EMBP][HIDP] bf16
                     const unsigned short* __restrict__ w2b,   // [HIDP][EMBP] bf16
                     const float* __restrict__ b1,
                     const float* __restrict__ b2,
                     float* hpre, int l) {
    __shared__ unsigned short zsh[32 * 648];  // z tile bf16, pitch 648 (1296B: 16B-aligned rows)
    __shared__ unsigned short ash[32 * 40];   // A k-tile staging, pitch 80B (16B-aligned)
    __shared__ unsigned short bsh[64 * 40];   // B k-tile staging, col-major, pitch 80B

    const int tid   = threadIdx.x;
    const int lane  = tid & 31;
    const int w     = tid >> 5;
    const int waveM = w >> 2;          // 0..1
    const int waveN = w & 3;           // 0..3
    const int m     = lane & 15;
    const int hi    = lane >> 4;       // 0 or 1
    const int row0  = blockIdx.x * 32;
    const float* b1l = b1 + (size_t)l * HID;
    const float* b2l = b2 + (size_t)l * EMB;

    // -------- Phase 1: z = relu(bf16(aggr) @ W1 + b1) -> zsh --------
    for (int nc = 0; nc < HIDP / 64; ++nc) {
        v8f acc = {};
        for (int kt = 0; kt < EMBP / 32; ++kt) {
            __syncthreads();
            for (int t = tid; t < 32 * 32; t += 256) {
                int r = t >> 5, k = t & 31;
                int kk = kt * 32 + k;
                float v = (kk < EMB) ? aggr[(size_t)(row0 + r) * EMB + kk] : 0.f;
                ash[r * 40 + k] = f2bf(v);
            }
            for (int t = tid; t < 32 * 64; t += 256) {
                int k = t >> 6, n = t & 63;   // n fast -> coalesced global read
                bsh[n * 40 + k] = w1b[(size_t)(kt * 32 + k) * HIDP + nc * 64 + n];
            }
            __syncthreads();
            Frag A, B;
            // A: 16x32 bf16; lane<16 holds K {0..7,16..23}, lane>=16 holds +8.
            // Two 16B runs per lane at dword (row*20 + hi*4) and +8.
            {
                const v4u* aq = (const v4u*)ash;
                int ai = (waveM * 16 + m) * 5 + hi;
                A.q[0] = aq[ai];
                A.q[1] = aq[ai + 2];
            }
            // B: 32x16; lane<16 holds K {0..7,16..23}, lane>=16 holds +8.
            {
                const v4u* bq = (const v4u*)bsh;
                int bi = (waveN * 16 + m) * 5 + hi;
                B.q[0] = bq[bi];
                B.q[1] = bq[bi + 2];
            }
            acc = __builtin_amdgcn_wmma_f32_16x16x32_bf16(
                      false, A.bf, false, B.bf, (short)0, acc, false, false);
        }
        int col = nc * 64 + waveN * 16 + m;
        float bias = (col < HID) ? b1l[col] : 0.f;   // one load per tile column
        #pragma unroll
        for (int r = 0; r < 8; ++r) {
            int row = waveM * 16 + r + (hi << 3);    // C/D: lanes>=16 are M+8
            float v = fmaxf(acc[r] + bias, 0.f);     // padded cols stay 0
            zsh[row * 648 + col] = f2bf(v);
        }
    }
    __syncthreads();

    // -------- Phase 2: hpre = z @ W2 + b2 --------
    for (int nc = 0; nc < EMBP / 64; ++nc) {
        v8f acc = {};
        for (int kt = 0; kt < HIDP / 32; ++kt) {
            __syncthreads();
            for (int t = tid; t < 32 * 64; t += 256) {
                int k = t >> 6, n = t & 63;
                bsh[n * 40 + k] = w2b[(size_t)(kt * 32 + k) * EMBP + nc * 64 + n];
            }
            __syncthreads();
            Frag A, B;
            {   // A from zsh: dword base row*324 + kt*16 + hi*4, runs of 4 dwords at +0 and +8
                const v4u* zq = (const v4u*)zsh;
                int zi = (waveM * 16 + m) * 81 + kt * 4 + hi;
                A.q[0] = zq[zi];
                A.q[1] = zq[zi + 2];
            }
            {
                const v4u* bq = (const v4u*)bsh;
                int bi = (waveN * 16 + m) * 5 + hi;
                B.q[0] = bq[bi];
                B.q[1] = bq[bi + 2];
            }
            acc = __builtin_amdgcn_wmma_f32_16x16x32_bf16(
                      false, A.bf, false, B.bf, (short)0, acc, false, false);
        }
        int col = nc * 64 + waveN * 16 + m;
        if (col < EMB) {
            float bias = b2l[col];                   // one load per tile column
            #pragma unroll
            for (int r = 0; r < 8; ++r) {
                int row = waveM * 16 + r + (hi << 3);
                hpre[(size_t)(row0 + row) * EMB + col] = acc[r] + bias;
            }
        }
    }
}

// ---------------------------------------------------------------- BatchNorm
__global__ void bn_zero_kernel(float* bnsum) {
    int i = blockIdx.x * blockDim.x + threadIdx.x;
    if (i < 2 * EMB) bnsum[i] = 0.f;
}

__global__ void bn_stats_kernel(const float* __restrict__ hpre, float* bnsum) {
    int c = blockIdx.x;                       // feature column
    int per = (NN + gridDim.y - 1) / gridDim.y;
    int i0 = blockIdx.y * per;
    int i1 = i0 + per; if (i1 > NN) i1 = NN;
    float s = 0.f, s2 = 0.f;
    for (int i = i0 + threadIdx.x; i < i1; i += blockDim.x) {
        float v = hpre[(size_t)i * EMB + c];
        s += v; s2 += v * v;
    }
    __shared__ float r1[256], r2[256];
    r1[threadIdx.x] = s; r2[threadIdx.x] = s2;
    __syncthreads();
    for (int o = 128; o > 0; o >>= 1) {
        if (threadIdx.x < o) {
            r1[threadIdx.x] += r1[threadIdx.x + o];
            r2[threadIdx.x] += r2[threadIdx.x + o];
        }
        __syncthreads();
    }
    if (threadIdx.x == 0) {
        atomicAdd(&bnsum[c], r1[0]);
        atomicAdd(&bnsum[EMB + c], r2[0]);
    }
}

__global__ void bn_finalize_kernel(const float* __restrict__ bnsum,
                                   const float* __restrict__ bng,
                                   const float* __restrict__ bnb,
                                   float* bnsc, int l) {
    int c = blockIdx.x * blockDim.x + threadIdx.x;
    if (c >= EMB) return;
    float mean = bnsum[c] * (1.f / (float)NN);
    float var  = bnsum[EMB + c] * (1.f / (float)NN) - mean * mean;
    float sc   = bng[(size_t)l * EMB + c] * rsqrtf(var + 1e-5f);
    bnsc[c]       = sc;
    bnsc[EMB + c] = bnb[(size_t)l * EMB + c] - mean * sc;
}

__global__ void bn_apply_kernel(const float* __restrict__ hpre,
                                const float* __restrict__ bnsc,
                                float* __restrict__ h, int relu) {
    int idx = blockIdx.x * blockDim.x + threadIdx.x;
    if (idx >= NN * EMB) return;
    int c = idx % EMB;
    float v = hpre[idx] * bnsc[c] + bnsc[EMB + c];
    if (relu) v = fmaxf(v, 0.f);
    h[idx] = v;
}

// ---------------------------------------------------------------- fragment pooling
__global__ void frag_zero_kernel(float* fragsum, float* fragcnt) {
    int i = blockIdx.x * blockDim.x + threadIdx.x;
    if (i < FF * EMB) fragsum[i] = 0.f;
    if (i < FF) fragcnt[i] = 0.f;
}

__global__ void frag_accum_kernel(const float* __restrict__ h,
                                  const int* __restrict__ fb,
                                  float* fragsum, float* fragcnt) {
    int idx = blockIdx.x * blockDim.x + threadIdx.x;
    if (idx >= NN * EMB) return;
    int i = idx / EMB, c = idx - i * EMB;
    int f = fb[i];
    atomicAdd(&fragsum[(size_t)f * EMB + c], h[idx]);
    if (c == 0) atomicAdd(&fragcnt[f], 1.f);
}

__global__ void build_out_kernel(const float* __restrict__ h,
                                 const int* __restrict__ didx,
                                 const int* __restrict__ fb,
                                 const float* __restrict__ fragsum,
                                 const float* __restrict__ fragcnt,
                                 float* __restrict__ outb) {
    int idx = blockIdx.x * blockDim.x + threadIdx.x;
    if (idx >= NDANG * 2 * EMB) return;
    int i = idx / (2 * EMB), c = idx - i * (2 * EMB);
    int node = didx[i];
    float v;
    if (c < EMB) {
        v = h[(size_t)node * EMB + c];
    } else {
        int f = fb[node];
        v = fragsum[(size_t)f * EMB + (c - EMB)] / fmaxf(fragcnt[f], 1.f);
    }
    outb[idx] = v * (1.f / (float)EMB);
}

// ---------------------------------------------------------------- edge head
__global__ void edge_mm_kernel(const float* __restrict__ outb,
                               const int* __restrict__ de,
                               const int* __restrict__ dea,
                               const float* __restrict__ E1,
                               const float* __restrict__ E2,
                               float* __restrict__ out0p) {
    int d = blockIdx.x;
    int n0 = de[d];
    int a0 = dea[2 * d], a1 = dea[2 * d + 1];
    __shared__ float sh[HID];
    for (int k = threadIdx.x; k < HID; k += blockDim.x)
        sh[k] = outb[(size_t)n0 * HID + k];
    __syncthreads();
    const float* e1 = E1 + (size_t)a0 * HID * HID;
    const float* e2 = E2 + (size_t)a1 * HID * HID;
    for (int mm = threadIdx.x; mm < HID; mm += blockDim.x) {
        float acc = 0.f;
        for (int k = 0; k < HID; ++k)
            acc = fmaf(sh[k], e1[(size_t)k * HID + mm] + e2[(size_t)k * HID + mm], acc);
        out0p[(size_t)d * HID + mm] = acc;
    }
}

__global__ void final_dots_kernel(const float* __restrict__ out0p,
                                  const float* __restrict__ outb,
                                  const int* __restrict__ de,
                                  float* __restrict__ logits) {
    int t = blockIdx.x * blockDim.x + threadIdx.x;
    if (t >= 2 * DD) return;
    int which = t / DD, d = t - which * DD;
    int dd = which ? ((d + DD - 1) % DD) : d;      // roll(out1, 1) for second half
    int n1 = de[DD + dd];
    const float* a = out0p + (size_t)d * HID;
    const float* b = outb + (size_t)n1 * HID;
    float acc = 0.f;
    for (int k = 0; k < HID; ++k) acc = fmaf(a[k], b[k], acc);
    logits[t] = acc;
}

// ---------------------------------------------------------------- launcher
extern "C" void kernel_launch(void* const* d_in, const int* in_sizes, int n_in,
                              void* d_out, int out_size, void* d_ws, size_t ws_size,
                              hipStream_t stream) {
    const int*   x    = (const int*)  d_in[0];
    const int*   ei   = (const int*)  d_in[1];
    const int*   ea   = (const int*)  d_in[2];
    const int*   didx = (const int*)  d_in[3];
    const int*   fb   = (const int*)  d_in[4];
    const int*   de   = (const int*)  d_in[5];
    const int*   dea  = (const int*)  d_in[6];
    const float* ae1  = (const float*)d_in[7];
    const float* ae2  = (const float*)d_in[8];
    const float* ge1  = (const float*)d_in[9];
    const float* ge2  = (const float*)d_in[10];
    const float* W1   = (const float*)d_in[11];
    const float* b1   = (const float*)d_in[12];
    const float* W2   = (const float*)d_in[13];
    const float* b2   = (const float*)d_in[14];
    const float* bng  = (const float*)d_in[15];
    const float* bnb  = (const float*)d_in[16];
    const float* E1   = (const float*)d_in[17];
    const float* E2   = (const float*)d_in[18];
    float* logits = (float*)d_out;

    char* p = (char*)d_ws;
    auto carve = [&](size_t bytes) -> void* {
        void* q = (void*)p;
        p += (bytes + 255) & ~(size_t)255;
        return q;
    };
    float*          h       = (float*)carve((size_t)NN * EMB * 4);
    float*          aggr    = (float*)carve((size_t)NN * EMB * 4);
    unsigned short* w1b     = (unsigned short*)carve((size_t)EMBP * HIDP * 2);
    unsigned short* w2b     = (unsigned short*)carve((size_t)HIDP * EMBP * 2);
    float*          bnsum   = (float*)carve(2 * EMB * 4);
    float*          bnsc    = (float*)carve(2 * EMB * 4);
    float*          fragsum = (float*)carve((size_t)FF * EMB * 4);
    float*          fragcnt = (float*)carve((size_t)FF * 4);
    float*          outb    = (float*)carve((size_t)NDANG * 2 * EMB * 4);
    float*          out0p   = (float*)carve((size_t)DD * HID * 4);

    const int T = 256;
    dim3 blk(T);
    int gNE = (NN * EMB + T - 1) / T;

    embed_kernel<<<gNE, blk, 0, stream>>>(x, ae1, ae2, h);

    for (int l = 0; l < NLAYERS; ++l) {
        conv_w_kernel<<<(2 * EMBP * HIDP + T - 1) / T, blk, 0, stream>>>(W1, W2, w1b, w2b, l);
        aggr_init_kernel<<<gNE, blk, 0, stream>>>(h, ge1, ge2, aggr, l);
        long long eTot = (long long)EE * EMB;
        scatter_kernel<<<(unsigned)((eTot + T - 1) / T), blk, 0, stream>>>(h, ei, ea, ge1, ge2, aggr, l);
        // fused MLP: reads aggr rows, writes pre-BN result back into aggr (block-local rows)
        mlp_wmma_kernel<<<NN / 32, blk, 0, stream>>>(aggr, w1b, w2b, b1, b2, aggr, l);
        bn_zero_kernel<<<(2 * EMB + T - 1) / T, blk, 0, stream>>>(bnsum);
        bn_stats_kernel<<<dim3(EMB, 16), blk, 0, stream>>>(aggr, bnsum);
        bn_finalize_kernel<<<(EMB + T - 1) / T, blk, 0, stream>>>(bnsum, bng, bnb, bnsc, l);
        bn_apply_kernel<<<gNE, blk, 0, stream>>>(aggr, bnsc, h, (l < NLAYERS - 1) ? 1 : 0);
    }

    frag_zero_kernel<<<(FF * EMB + T - 1) / T, blk, 0, stream>>>(fragsum, fragcnt);
    frag_accum_kernel<<<gNE, blk, 0, stream>>>(h, fb, fragsum, fragcnt);
    build_out_kernel<<<(NDANG * 2 * EMB + T - 1) / T, blk, 0, stream>>>(h, didx, fb, fragsum, fragcnt, outb);
    edge_mm_kernel<<<DD, blk, 0, stream>>>(outb, de, dea, E1, E2, out0p);
    final_dots_kernel<<<(2 * DD + T - 1) / T, blk, 0, stream>>>(out0p, outb, de, logits);
}
